// NeuralNetwork_50568944943407
// MI455X (gfx1250) — matrix operand order repaired and checked by reading, not compiled, not run on hardware
//
#include <hip/hip_runtime.h>
#include <hip/hip_bf16.h>

typedef __attribute__((ext_vector_type(16))) _Float16 v16h;
typedef __attribute__((ext_vector_type(8)))  _Float16 v8h;
typedef __attribute__((ext_vector_type(4)))  _Float16 v4h;
typedef __attribute__((ext_vector_type(8)))  float    v8f;

#define TLEN 8192
#define BSZ  128
#define TILE 128
#define NW   8
#define NTHREADS 256

#define NCH  128   // Kalman chunks per row
#define CLEN 64    // chunk length (NCH*CLEN == TLEN)

// ---------------------------------------------------------------------------
// Fused conv pipeline. Position-major feature layout feat[pos*CI + ci] with
// tap-major K ordering (kk = tap*CI + ci) makes the im2col A fragment two
// contiguous 16B chunks: addr(m,kk) = (sj+m)*CI + kk. Weights padded to
// [CO][KPAD] (zero-filled) make B the same two 16B chunks. No branches, no
// per-element index math in the WMMA inner loop.
//
// Buffers (positions relative to tile start t0):
//   x : [-8, 138]  147 staged + 1 pad, CI_eff=4 (ch 2,3 zero)
//   h1: [-6, 133]  140 computed + 1 zero pad pos, 16 ch
//   h2: [-4, 131]  136 positions, 32 ch
//   h3: [-2, 129]  132 positions, 64 ch
//   r : [ 0, 127]  written to global
// ---------------------------------------------------------------------------
struct __align__(16) ConvLDS {
  _Float16 x [148 * 4];    // 592 halves
  _Float16 h1[141 * 16];   // 2256
  _Float16 h2[136 * 32];   // 4352
  _Float16 h3[132 * 64];   // 8448
  _Float16 w1[16 * 32];    // [16][KPAD=32],  kk = tap*4  + ci (ci<2,tap<5 real)
  _Float16 w2[32 * 96];    // [32][KPAD=96],  kk = tap*16 + ci (kk<80 real)
  _Float16 w3[64 * 160];   // [64][KPAD=160], kk = tap*32 + ci (exact)
  _Float16 w4[2 * 320];    // [2][KPAD=320],  row 0 real, row 1 zeros
  float b1[16]; float b2[32]; float b3[64]; float b4[1];
};

template<int CI, int CO, int WOUT, int KPAD, int LAYER>
__device__ __forceinline__ void conv_layer(const _Float16* __restrict__ in,
                                           _Float16* __restrict__ out,
                                           const _Float16* __restrict__ w,
                                           const float* __restrict__ bias,
                                           int lane, int wv,
                                           float* __restrict__ r_out, int row_t0) {
  constexpr int KITERS = KPAD / 32;
  constexpr int NT     = (WOUT + 15) / 16;   // 16-wide time tiles (last overlaps)
  constexpr int NG     = (CO + 15) / 16;     // 16-wide channel groups
  const int hl = lane >> 4;
  const int mn = lane & 15;                  // A row (m) / B,D column (n)

  for (int task = wv; task < NG * NT; task += NW) {
    const int cg = task / NT;
    const int j  = task % NT;
    const int sj = (16 * j < WOUT - 16) ? 16 * j : (WOUT - 16);
    int cobase;
    if constexpr (LAYER == 4) cobase = (mn != 0) ? KPAD : 0;   // zero row for n!=0
    else                      cobase = (cg * 16 + mn) * KPAD;

    const _Float16* aptr = in + (sj + mn) * CI + 8 * hl;
    const _Float16* bptr = w + cobase + 8 * hl;
    v8f c = {0.f, 0.f, 0.f, 0.f, 0.f, 0.f, 0.f, 0.f};

#pragma unroll
    for (int ko = 0; ko < KITERS; ++ko) {
      const _Float16* ap = aptr + ko * 32;
      const _Float16* bp = bptr + ko * 32;
      // A fragment: lane holds K {8hl+0..7, 8hl+16..23} of a 32-wide K slice.
      v4h a0 = *(const v4h*)(ap);
      v4h a1 = *(const v4h*)(ap + 4);
      v4h a2 = *(const v4h*)(ap + 16);
      v4h a3 = *(const v4h*)(ap + 20);
      // B fragment: lane holds K {8hl+0..7, 8hl+16..23} for column n.
      v8h b0 = *(const v8h*)(bp);
      v8h b1 = *(const v8h*)(bp + 16);
      v16h A, Bv;
#pragma unroll
      for (int i = 0; i < 4; ++i) {
        A[i] = a0[i]; A[4 + i] = a1[i]; A[8 + i] = a2[i]; A[12 + i] = a3[i];
      }
#pragma unroll
      for (int i = 0; i < 8; ++i) { Bv[i] = b0[i]; Bv[8 + i] = b1[i]; }
      c = __builtin_amdgcn_wmma_f32_16x16x32_f16(false, A, false, Bv,
                                                 (short)0, c, false, false);
    }

    if constexpr (LAYER < 4) {
      const int co = cg * 16 + mn;
      const float bb = bias[co];
#pragma unroll
      for (int q = 0; q < 8; ++q) {
        int mm = q + 8 * hl;                  // D: VGPR q holds M = q (+8 upper half)
        float val = c[q] + bb;
        val = val > 0.f ? val : 0.f;
        out[(sj + mm) * CO + co] = (_Float16)val;
      }
    } else {
      if (mn == 0) {
        const float bb = bias[0];
#pragma unroll
        for (int q = 0; q < 8; ++q) {
          int mm = q + 8 * hl;
          float val = c[q] + bb;
          val = val > 0.f ? val : 0.f;
          r_out[row_t0 + sj + mm] = val;
        }
      }
    }
  }
}

// Reflect fixup at row edges (position-major): halo positions outside [0,T-1]
// replaced by in-tile reflections, matching JAX per-layer reflect padding.
template<int CO, int WOUT, int BASE>  // BASE < 0, halo H = -BASE each side
__device__ __forceinline__ void edge_fixup(_Float16* __restrict__ out, int t0, int tid) {
  constexpr int H = -BASE;
  if (t0 == 0) {
    for (int idx = tid; idx < H * CO; idx += NTHREADS) {
      int i = idx / CO, ch = idx % CO;                   // i in [0,H)
      out[i * CO + ch] = out[(-2 * BASE - i) * CO + ch];
    }
  }
  if (t0 == TLEN - TILE) {
    for (int idx = tid; idx < H * CO; idx += NTHREADS) {
      int ii = idx / CO, ch = idx % CO;
      int i = WOUT - H + ii;                             // OOB region [WOUT-H, WOUT)
      out[i * CO + ch] = out[(254 - 2 * BASE - i) * CO + ch];
    }
  }
}

__global__ void __launch_bounds__(NTHREADS)
conv_kernel(const float* __restrict__ a, const float* __restrict__ v,
            const float* __restrict__ w1, const float* __restrict__ b1,
            const float* __restrict__ w2, const float* __restrict__ b2,
            const float* __restrict__ w3, const float* __restrict__ b3,
            const float* __restrict__ w4, const float* __restrict__ b4,
            float* __restrict__ r_out) {
  __shared__ ConvLDS s;
  const int t0  = blockIdx.x * TILE;
  const int b   = blockIdx.y;
  const int tid = threadIdx.x;
  const int lane = tid & 31, wv = tid >> 5;

  // Zero entire LDS block (covers all K/channel/tail padding with finite zeros).
  {
    unsigned* p = (unsigned*)&s;
    constexpr int N = sizeof(ConvLDS) / 4;
    for (int i = tid; i < N; i += NTHREADS) p[i] = 0u;
  }
  __syncthreads();

  // Stage input (147 positions, global reflect) into pos-major x, ch 0/1.
  for (int i = tid; i < 147; i += NTHREADS) {
    int g = t0 - 8 + i;
    if (g < 0) g = -g;
    if (g > TLEN - 1) g = 2 * (TLEN - 1) - g;
    s.x[i * 4 + 0] = (_Float16)a[b * TLEN + g];
    s.x[i * 4 + 1] = (_Float16)v[b * TLEN + g];
  }
  // Stage weights tap-major into zero-padded [CO][KPAD] tables.
  for (int i = tid; i < 160; i += NTHREADS) {          // w1 [16][2][5]
    int co = i / 10, rem = i % 10, ci = rem / 5, tap = rem % 5;
    s.w1[co * 32 + tap * 4 + ci] = (_Float16)w1[i];
  }
  for (int i = tid; i < 2560; i += NTHREADS) {         // w2 [32][16][5]
    int co = i / 80, rem = i % 80, ci = rem / 5, tap = rem % 5;
    s.w2[co * 96 + tap * 16 + ci] = (_Float16)w2[i];
  }
  for (int i = tid; i < 10240; i += NTHREADS) {        // w3 [64][32][5]
    int co = i / 160, rem = i % 160, ci = rem / 5, tap = rem % 5;
    s.w3[co * 160 + tap * 32 + ci] = (_Float16)w3[i];
  }
  for (int i = tid; i < 320; i += NTHREADS) {          // w4 [1][64][5] -> row 0
    int ci = i / 5, tap = i % 5;
    s.w4[tap * 64 + ci] = (_Float16)w4[i];
  }
  for (int i = tid; i < 16; i += NTHREADS) s.b1[i] = b1[i];
  for (int i = tid; i < 32; i += NTHREADS) s.b2[i] = b2[i];
  for (int i = tid; i < 64; i += NTHREADS) s.b3[i] = b3[i];
  if (tid == 0) s.b4[0] = b4[0];
  __syncthreads();

  conv_layer<4, 16, 140, 32, 1>(s.x, s.h1, s.w1, s.b1, lane, wv, nullptr, 0);
  __syncthreads();
  edge_fixup<16, 140, -6>(s.h1, t0, tid);
  __syncthreads();
  conv_layer<16, 32, 136, 96, 2>(s.h1, s.h2, s.w2, s.b2, lane, wv, nullptr, 0);
  __syncthreads();
  edge_fixup<32, 136, -4>(s.h2, t0, tid);
  __syncthreads();
  conv_layer<32, 64, 132, 160, 3>(s.h2, s.h3, s.w3, s.b3, lane, wv, nullptr, 0);
  __syncthreads();
  edge_fixup<64, 132, -2>(s.h3, t0, tid);
  __syncthreads();
  conv_layer<64, 1, 128, 320, 4>(s.h3, nullptr, s.w4, s.b4, lane, wv,
                                 r_out, b * TLEN + t0);
}

// ---------------------------------------------------------------------------
// Kalman filter as a chunked associative scan (Moebius for P, affine for x).
// ---------------------------------------------------------------------------
__global__ void __launch_bounds__(NCH)
scanA_kernel(const float* __restrict__ r, const float* __restrict__ qp,
             float* __restrict__ gmat) {
  const int row = blockIdx.x, c = threadIdx.x;
  const float Q = *qp;
  const float* rr = r + row * TLEN + c * CLEN;
  float g00 = 1.f, g01 = 0.f, g10 = 0.f, g11 = 1.f;
  for (int i = 0; i < CLEN; ++i) {
    float R = rr[i];
    float n00 = R * g00 + Q * R * g10;
    float n01 = R * g01 + Q * R * g11;
    float n10 = g00 + (Q + R) * g10;
    float n11 = g01 + (Q + R) * g11;
    g00 = n00; g01 = n01; g10 = n10; g11 = n11;
    if ((i & 7) == 7) {  // periodic renorm (only ratios matter)
      float sc = 1.0f / (fabsf(g00) + fabsf(g01) + fabsf(g10) + fabsf(g11) + 1e-30f);
      g00 *= sc; g01 *= sc; g10 *= sc; g11 *= sc;
    }
  }
  float* o = gmat + (row * NCH + c) * 4;
  o[0] = g00; o[1] = g01; o[2] = g10; o[3] = g11;
}

__global__ void __launch_bounds__(BSZ)
scanB_kernel(const float* __restrict__ gmat, float* __restrict__ Pin) {
  const int row = threadIdx.x;
  float P = 0.f;
  for (int c = 0; c < NCH; ++c) {
    Pin[row * NCH + c] = P;
    const float* g = gmat + (row * NCH + c) * 4;
    P = (g[0] * P + g[1]) / (g[2] * P + g[3]);
  }
}

__global__ void __launch_bounds__(NCH)
scanC_kernel(const float* __restrict__ r, const float* __restrict__ a,
             const float* __restrict__ v, const float* __restrict__ qp,
             const float* __restrict__ mean, const float* __restrict__ sd,
             const float* __restrict__ Pin, float* __restrict__ AB) {
  const int row = blockIdx.x, c = threadIdx.x;
  const float Q = *qp, m0 = mean[0], m1 = mean[1], s0 = sd[0], s1 = sd[1];
  const int base = row * TLEN + c * CLEN;
  float P = Pin[row * NCH + c];
  float Ac = 1.f, Bc = 0.f;
  for (int i = 0; i < CLEN; ++i) {
    float R  = r[base + i];
    float ad = a[base + i] * s0 + m0;
    float z  = v[base + i] * s1 + m1;
    float Pp = P + Q;
    float K  = Pp / (Pp + R);
    float al = 1.f - K;
    float be = al * ad + K * z;
    Ac = al * Ac;
    Bc = al * Bc + be;
    P  = al * Pp;
  }
  AB[(row * NCH + c) * 2]     = Ac;
  AB[(row * NCH + c) * 2 + 1] = Bc;
}

__global__ void __launch_bounds__(BSZ)
scanD_kernel(const float* __restrict__ AB, float* __restrict__ Xin) {
  const int row = threadIdx.x;
  float x = 0.f;
  for (int c = 0; c < NCH; ++c) {
    Xin[row * NCH + c] = x;
    x = AB[(row * NCH + c) * 2] * x + AB[(row * NCH + c) * 2 + 1];
  }
}

__global__ void __launch_bounds__(NCH)
scanE_kernel(const float* __restrict__ r, const float* __restrict__ a,
             const float* __restrict__ v, const float* __restrict__ qp,
             const float* __restrict__ mean, const float* __restrict__ sd,
             const float* __restrict__ Pin, const float* __restrict__ Xin,
             float* __restrict__ y) {
  const int row = blockIdx.x, c = threadIdx.x;
  const float Q = *qp, m0 = mean[0], m1 = mean[1], s0 = sd[0], s1 = sd[1];
  const int base = row * TLEN + c * CLEN;
  float P = Pin[row * NCH + c];
  float x = Xin[row * NCH + c];
  for (int i = 0; i < CLEN; ++i) {
    float R  = r[base + i];
    float ad = a[base + i] * s0 + m0;
    float z  = v[base + i] * s1 + m1;
    float Pp = P + Q;
    float K  = Pp / (Pp + R);
    float al = 1.f - K;
    x = al * (x + ad) + K * z;
    P = al * Pp;
    y[base + i] = x;
  }
}

extern "C" void kernel_launch(void* const* d_in, const int* in_sizes, int n_in,
                              void* d_out, int out_size, void* d_ws, size_t ws_size,
                              hipStream_t stream) {
  const float* a    = (const float*)d_in[0];
  const float* v    = (const float*)d_in[1];
  const float* w1   = (const float*)d_in[2];
  const float* b1   = (const float*)d_in[3];
  const float* w2   = (const float*)d_in[4];
  const float* b2   = (const float*)d_in[5];
  const float* w3   = (const float*)d_in[6];
  const float* b3   = (const float*)d_in[7];
  const float* w4   = (const float*)d_in[8];
  const float* b4   = (const float*)d_in[9];
  const float* Q    = (const float*)d_in[10];
  const float* mean = (const float*)d_in[11];
  const float* sd   = (const float*)d_in[12];

  float* y_out = (float*)d_out;                 // [B, T]
  float* r_out = (float*)d_out + BSZ * TLEN;    // [B, T] (r[:,0,:])

  float* ws   = (float*)d_ws;
  float* gmat = ws;                              // B*NCH*4
  float* Pin  = gmat + BSZ * NCH * 4;            // B*NCH
  float* AB   = Pin  + BSZ * NCH;                // B*NCH*2
  float* Xin  = AB   + BSZ * NCH * 2;            // B*NCH

  conv_kernel<<<dim3(TLEN / TILE, BSZ), NTHREADS, 0, stream>>>(
      a, v, w1, b1, w2, b2, w3, b3, w4, b4, r_out);

  scanA_kernel<<<BSZ, NCH, 0, stream>>>(r_out, Q, gmat);
  scanB_kernel<<<1, BSZ, 0, stream>>>(gmat, Pin);
  scanC_kernel<<<BSZ, NCH, 0, stream>>>(r_out, a, v, Q, mean, sd, Pin, AB);
  scanD_kernel<<<1, BSZ, 0, stream>>>(AB, Xin);
  scanE_kernel<<<BSZ, NCH, 0, stream>>>(r_out, a, v, Q, mean, sd, Pin, Xin, y_out);
}
